// SequentialGPTOSSMoEV1_16604343566460
// MI455X (gfx1250) — matrix operand order repaired and checked by reading, not compiled, not run on hardware
//
#include <hip/hip_runtime.h>
#include <hip/hip_bf16.h>

// MoE (GPT-OSS style) for MI455X / gfx1250.
// Top-k-only compute (4096 token-expert pairs), bf16 WMMA (v_wmma_f32_16x16x32_bf16),
// f32 accumulation, expert-major tiling for L2 reuse (each expert's ~100MB fp32
// weight slice fits the 192MB L2 while its token tiles run).
// GEMM loops are double-buffered: next chunk's weight tiles staged via registers,
// next chunk's bf16 tiles staged via global_load_async_to_lds_b128 (ASYNCcnt),
// one barrier per K-chunk. f32->bf16 via v_cvt_pk_bf16_f32 (1 VALU op / pair).
//
// Workspace layout (bytes), total ~29.6 MB:
//   [0,32)        counts[8]
//   [32,64)       offsets[8]
//   [64,32832)    tok_ids[8][1024]
//   [33024, +5.9M)   x in bf16 (1024 x 2880)
//   [+, +23.6M)      act in bf16 (4096 pairs x 2880)

#define H_DIM 2880
#define I_DIM 2880
#define E_NUM 8
#define N_TOK 1024
#define TOPK  4
#define ALPHA_C 1.702f
#define LIMIT_C 7.0f

typedef __attribute__((ext_vector_type(16))) __bf16 v16bf;
typedef __attribute__((ext_vector_type(8)))  float  v8f;

union Frag16 {
    uint4 u[2];
    v16bf v;
};

// f32 -> bf16 (RNE), packed pair in one VALU op.
__device__ __forceinline__ unsigned int pack_bf16x2(float lo, float hi) {
#if __has_builtin(__builtin_amdgcn_cvt_pk_bf16_f32)
    typedef __bf16 bf16x2 __attribute__((ext_vector_type(2)));
    bf16x2 p = __builtin_amdgcn_cvt_pk_bf16_f32(lo, hi);
    return __builtin_bit_cast(unsigned int, p);
#else
    unsigned int r;
    asm("v_cvt_pk_bf16_f32 %0, %1, %2" : "=v"(r) : "v"(lo), "v"(hi));
    return r;
#endif
}

__device__ __forceinline__ unsigned short f32_to_bf16(float f) {
    return (unsigned short)(pack_bf16x2(f, 0.f) & 0xFFFFu);
}

// CDNA5 async memory->LDS copy (16B per lane), tracked by ASYNCcnt.
// Generic pointers to __shared__ carry the LDS offset in their low 32 bits.
__device__ __forceinline__ void async_copy_b128(void* lds_dst, const void* gsrc) {
    unsigned lds_off = (unsigned)(size_t)lds_dst;
    unsigned long long ga = (unsigned long long)(size_t)gsrc;
    asm volatile("global_load_async_to_lds_b128 %0, %1, off"
                 :: "v"(lds_off), "v"(ga) : "memory");
}
__device__ __forceinline__ void wait_async0() {
    asm volatile("s_wait_asynccnt 0x0" ::: "memory");
}

// ---------------------------------------------------------------------------
// 1) Router: one wave per token. logits -> top-4 -> softmax -> scores.
//    Also converts the token's hidden state to bf16 for the WMMA GEMMs.
// ---------------------------------------------------------------------------
__global__ __launch_bounds__(256)
void router_kernel(const float* __restrict__ x, const float* __restrict__ Wr,
                   const float* __restrict__ br, float* __restrict__ scores,
                   unsigned short* __restrict__ xb)
{
    int gtid = blockIdx.x * 256 + threadIdx.x;
    int n    = gtid >> 5;
    int lane = gtid & 31;
    if (n >= N_TOK) return;

    float acc[E_NUM];
#pragma unroll
    for (int e = 0; e < E_NUM; ++e) acc[e] = 0.f;

    for (int h = lane; h < H_DIM; h += 32) {
        float xv = x[(size_t)n * H_DIM + h];
        xb[(size_t)n * H_DIM + h] = f32_to_bf16(xv);
#pragma unroll
        for (int e = 0; e < E_NUM; ++e) acc[e] += xv * Wr[e * H_DIM + h];
    }
#pragma unroll
    for (int e = 0; e < E_NUM; ++e) {
#pragma unroll
        for (int off = 16; off > 0; off >>= 1)
            acc[e] += __shfl_xor(acc[e], off, 32);
    }
    if (lane == 0) {
        float logit[E_NUM];
#pragma unroll
        for (int e = 0; e < E_NUM; ++e) logit[e] = acc[e] + br[e];

        unsigned used = 0;
        float tv[TOPK]; int ti[TOPK];
#pragma unroll
        for (int k = 0; k < TOPK; ++k) {
            float best = -3.4e38f; int bi = 0;
#pragma unroll
            for (int e = 0; e < E_NUM; ++e)
                if (!((used >> e) & 1u) && logit[e] > best) { best = logit[e]; bi = e; }
            tv[k] = best; ti[k] = bi; used |= (1u << bi);
        }
        float mx = tv[0], s = 0.f, w[TOPK];
#pragma unroll
        for (int k = 0; k < TOPK; ++k) { w[k] = __expf(tv[k] - mx); s += w[k]; }
        float inv = __builtin_amdgcn_rcpf(s);
        float outp[E_NUM];
#pragma unroll
        for (int e = 0; e < E_NUM; ++e) outp[e] = 0.f;
#pragma unroll
        for (int k = 0; k < TOPK; ++k) outp[ti[k]] = w[k] * inv;
#pragma unroll
        for (int e = 0; e < E_NUM; ++e) scores[n * E_NUM + e] = outp[e];
    }
}

// ---------------------------------------------------------------------------
// 2) Deterministic per-expert compaction: one wave per expert, ordered ballot.
// ---------------------------------------------------------------------------
__global__ void compact_kernel(const float* __restrict__ scores,
                               int* __restrict__ counts, int* __restrict__ tok_ids)
{
    int e = blockIdx.x;
    int lane = threadIdx.x;
    int cnt = 0;
    for (int base = 0; base < N_TOK; base += 32) {
        int n = base + lane;
        bool sel = scores[n * E_NUM + e] > 0.f;
        unsigned mask = __builtin_amdgcn_ballot_w32(sel);
        int pre = __popc(mask & ((1u << lane) - 1u));
        if (sel) tok_ids[e * N_TOK + cnt + pre] = n;
        cnt += __popc(mask);
    }
    if (lane == 0) counts[e] = cnt;
}

__global__ void offsets_kernel(const int* __restrict__ counts, int* __restrict__ offsets)
{
    if (threadIdx.x == 0) {
        int o = 0;
        for (int e = 0; e < E_NUM; ++e) { offsets[e] = o; o += counts[e]; }
    }
}

// ---------------------------------------------------------------------------
// 3) out[n,h] = sum_e score[n,e] * bd[e,h]   (down-proj bias, score-weighted)
// ---------------------------------------------------------------------------
__global__ __launch_bounds__(256)
void bias_init_kernel(const float* __restrict__ bd, const float* __restrict__ scores,
                      float* __restrict__ out)
{
    int idx = blockIdx.x * 256 + threadIdx.x;
    if (idx >= N_TOK * H_DIM) return;
    int n = idx / H_DIM;
    int h = idx - n * H_DIM;
    float s = 0.f;
#pragma unroll
    for (int e = 0; e < E_NUM; ++e) s += scores[n * E_NUM + e] * bd[e * H_DIM + h];
    out[idx] = s;
}

// ---------------------------------------------------------------------------
// 4) Gate+Up GEMM (shared A fragment) + fused clamp/GLU epilogue -> bf16 act.
//    64 tokens x 64 I-cols per block; 8 waves x (16x32 strip of gate AND up).
//    Double-buffered K (chunk=32): x tile via async-to-LDS, weight tiles via
//    register staging (load f32 before the WMMAs, pack+ds_store after).
// ---------------------------------------------------------------------------
__global__ __launch_bounds__(256)
void moe_gateup_kernel(const unsigned short* __restrict__ xb,
                       const float* __restrict__ Wg, const float* __restrict__ bg,
                       const float* __restrict__ Wu, const float* __restrict__ bu,
                       const int* __restrict__ counts, const int* __restrict__ offsets,
                       const int* __restrict__ tok_ids, unsigned short* __restrict__ act)
{
    const int e    = blockIdx.z;
    const int tile = blockIdx.y;
    const int cnt  = counts[e];
    if (tile * 64 >= cnt) return;
    const int ibase = blockIdx.x * 64;

    __shared__ __align__(16) unsigned short xs[2][64 * 32];
    __shared__ __align__(16) unsigned short gs[2][64 * 32];
    __shared__ __align__(16) unsigned short us[2][64 * 32];
    __shared__ int toks[64];

    const int t = threadIdx.x;
    if (t < 64) {
        int row = tile * 64 + t;
        toks[t] = tok_ids[e * N_TOK + (row < cnt ? row : cnt - 1)];
    }
    __syncthreads();

    const int lane = t & 31;
    const int wave = t >> 5;
    const int mi   = wave & 3;   // 4 row-blocks of 16
    const int nh   = wave >> 2;  // 2 col-halves of 32

    v8f ag0 = {0,0,0,0,0,0,0,0};
    v8f ag1 = ag0, au0 = ag0, au1 = ag0;

    const int m_ld = t >> 2;          // staging row 0..63
    const int kk   = (t & 3) * 8;     // staging k-offset 0/8/16/24
    const size_t wbase = (size_t)e * I_DIM * H_DIM;
    const int nl  = lane & 15;
    const int kbA = (lane < 16) ? 0 : 16;  // A 16-bit, lane-half => K+16 (contiguous, matches B)
    const int kbB = (lane < 16) ? 0 : 16;  // B 16-bit, lane-half => K+16 (contiguous, matches A)
    const int mrow = mi * 16 + nl;
    const int nb0 = nh * 32, nb1 = nh * 32 + 16;

    const unsigned short* xrow = xb + (size_t)toks[m_ld] * H_DIM;
    const float* grow = Wg + wbase + (size_t)(ibase + m_ld) * H_DIM;
    const float* urow = Wu + wbase + (size_t)(ibase + m_ld) * H_DIM;

    // -------- prologue: stage chunk 0 into buffer 0 --------
    async_copy_b128(&xs[0][m_ld * 32 + kk], xrow + kk);
    {
        const float4* gp = reinterpret_cast<const float4*>(grow + kk);
        float4 a = gp[0], b = gp[1];
        uint4 p;
        p.x = pack_bf16x2(a.x, a.y); p.y = pack_bf16x2(a.z, a.w);
        p.z = pack_bf16x2(b.x, b.y); p.w = pack_bf16x2(b.z, b.w);
        *reinterpret_cast<uint4*>(&gs[0][m_ld * 32 + kk]) = p;
        const float4* up = reinterpret_cast<const float4*>(urow + kk);
        a = up[0]; b = up[1];
        p.x = pack_bf16x2(a.x, a.y); p.y = pack_bf16x2(a.z, a.w);
        p.z = pack_bf16x2(b.x, b.y); p.w = pack_bf16x2(b.z, b.w);
        *reinterpret_cast<uint4*>(&us[0][m_ld * 32 + kk]) = p;
    }
    wait_async0();
    __syncthreads();

    int it = 0;
    for (int k0 = 0; k0 < H_DIM; k0 += 32, it ^= 1) {
        const int kn = k0 + 32;
        const bool more = (kn < H_DIM);
        float4 g0 = {}, g1 = {}, u0 = {}, u1 = {};
        if (more) {
            // issue next chunk's loads before this chunk's WMMAs
            async_copy_b128(&xs[it ^ 1][m_ld * 32 + kk], xrow + kn + kk);
            const float4* gp = reinterpret_cast<const float4*>(grow + kn + kk);
            g0 = gp[0]; g1 = gp[1];
            const float4* up = reinterpret_cast<const float4*>(urow + kn + kk);
            u0 = up[0]; u1 = up[1];
        }

        const unsigned short* xsb = xs[it];
        const unsigned short* gsb = gs[it];
        const unsigned short* usb = us[it];
        Frag16 fa, fg0, fg1, fu0, fu1;
        fa.u[0]  = *reinterpret_cast<const uint4*>(&xsb[mrow * 32 + kbA]);
        fa.u[1]  = *reinterpret_cast<const uint4*>(&xsb[mrow * 32 + kbA + 8]);
        fg0.u[0] = *reinterpret_cast<const uint4*>(&gsb[(nb0 + nl) * 32 + kbB]);
        fg0.u[1] = *reinterpret_cast<const uint4*>(&gsb[(nb0 + nl) * 32 + kbB + 8]);
        fg1.u[0] = *reinterpret_cast<const uint4*>(&gsb[(nb1 + nl) * 32 + kbB]);
        fg1.u[1] = *reinterpret_cast<const uint4*>(&gsb[(nb1 + nl) * 32 + kbB + 8]);
        fu0.u[0] = *reinterpret_cast<const uint4*>(&usb[(nb0 + nl) * 32 + kbB]);
        fu0.u[1] = *reinterpret_cast<const uint4*>(&usb[(nb0 + nl) * 32 + kbB + 8]);
        fu1.u[0] = *reinterpret_cast<const uint4*>(&usb[(nb1 + nl) * 32 + kbB]);
        fu1.u[1] = *reinterpret_cast<const uint4*>(&usb[(nb1 + nl) * 32 + kbB + 8]);

        ag0 = __builtin_amdgcn_wmma_f32_16x16x32_bf16(false, fa.v, false, fg0.v, (short)0, ag0, false, false);
        ag1 = __builtin_amdgcn_wmma_f32_16x16x32_bf16(false, fa.v, false, fg1.v, (short)0, ag1, false, false);
        au0 = __builtin_amdgcn_wmma_f32_16x16x32_bf16(false, fa.v, false, fu0.v, (short)0, au0, false, false);
        au1 = __builtin_amdgcn_wmma_f32_16x16x32_bf16(false, fa.v, false, fu1.v, (short)0, au1, false, false);

        if (more) {
            uint4 p;
            p.x = pack_bf16x2(g0.x, g0.y); p.y = pack_bf16x2(g0.z, g0.w);
            p.z = pack_bf16x2(g1.x, g1.y); p.w = pack_bf16x2(g1.z, g1.w);
            *reinterpret_cast<uint4*>(&gs[it ^ 1][m_ld * 32 + kk]) = p;
            p.x = pack_bf16x2(u0.x, u0.y); p.y = pack_bf16x2(u0.z, u0.w);
            p.z = pack_bf16x2(u1.x, u1.y); p.w = pack_bf16x2(u1.z, u1.w);
            *reinterpret_cast<uint4*>(&us[it ^ 1][m_ld * 32 + kk]) = p;
            wait_async0();
        }
        __syncthreads();
    }

    // Epilogue: D layout => lane = N (+16 for M>=8), VGPR r = M row.
    const int mb    = (lane < 16) ? 0 : 8;
    const int pbase = offsets[e] + tile * 64;
#pragma unroll
    for (int half = 0; half < 2; ++half) {
        const v8f& g = half ? ag1 : ag0;
        const v8f& u = half ? au1 : au0;
        int ig = ibase + nh * 32 + half * 16 + nl;
        float bgv = bg[e * I_DIM + ig];
        float buv = bu[e * I_DIM + ig];
#pragma unroll
        for (int r = 0; r < 8; ++r) {
            int ml = mi * 16 + mb + r;
            if (tile * 64 + ml < cnt) {
                float gv = g[r] + bgv;
                gv = fminf(gv, LIMIT_C);
                float uv = u[r] + buv;
                uv = fminf(fmaxf(uv, -LIMIT_C), LIMIT_C);
                float glu = gv * __builtin_amdgcn_rcpf(1.f + __expf(-ALPHA_C * gv));
                float av  = (uv + 1.f) * glu;
                act[(size_t)(pbase + ml) * I_DIM + ig] = f32_to_bf16(av);
            }
        }
    }
}

// ---------------------------------------------------------------------------
// 5) Down projection: y = act @ Wd[e]^T ; score-weighted atomic add into out.
//    Same double-buffered pipeline: act tile async-to-LDS, Wd via registers.
// ---------------------------------------------------------------------------
__global__ __launch_bounds__(256)
void moe_down_kernel(const unsigned short* __restrict__ act,
                     const float* __restrict__ Wd,
                     const int* __restrict__ counts, const int* __restrict__ offsets,
                     const int* __restrict__ tok_ids, const float* __restrict__ scores,
                     float* __restrict__ out)
{
    const int e    = blockIdx.z;
    const int tile = blockIdx.y;
    const int cnt  = counts[e];
    if (tile * 64 >= cnt) return;
    const int hbase = blockIdx.x * 64;

    __shared__ __align__(16) unsigned short as_[2][64 * 32];
    __shared__ __align__(16) unsigned short ds_[2][64 * 32];
    __shared__ int toks[64];

    const int t = threadIdx.x;
    const int pbase = offsets[e] + tile * 64;
    if (t < 64) {
        int row = tile * 64 + t;
        toks[t] = tok_ids[e * N_TOK + (row < cnt ? row : cnt - 1)];
    }
    __syncthreads();

    const int lane = t & 31;
    const int wave = t >> 5;
    const int mi   = wave & 3;
    const int nh   = wave >> 2;

    v8f a0 = {0,0,0,0,0,0,0,0};
    v8f a1 = a0;

    const int m_ld = t >> 2;
    const int kk   = (t & 3) * 8;
    const int nl   = lane & 15;
    const int kbA  = (lane < 16) ? 0 : 16;
    const int kbB  = (lane < 16) ? 0 : 16;
    const int mrow = mi * 16 + nl;
    const int nb0 = nh * 32, nb1 = nh * 32 + 16;
    const int rem  = cnt - 1 - tile * 64;
    const int arow = pbase + (m_ld < rem ? m_ld : rem);

    const unsigned short* acrow = act + (size_t)arow * I_DIM;
    const float* wrow = Wd + ((size_t)e * H_DIM + (hbase + m_ld)) * I_DIM;

    // -------- prologue: stage chunk 0 into buffer 0 --------
    async_copy_b128(&as_[0][m_ld * 32 + kk], acrow + kk);
    {
        const float4* s = reinterpret_cast<const float4*>(wrow + kk);
        float4 a = s[0], b = s[1];
        uint4 p;
        p.x = pack_bf16x2(a.x, a.y); p.y = pack_bf16x2(a.z, a.w);
        p.z = pack_bf16x2(b.x, b.y); p.w = pack_bf16x2(b.z, b.w);
        *reinterpret_cast<uint4*>(&ds_[0][m_ld * 32 + kk]) = p;
    }
    wait_async0();
    __syncthreads();

    int it = 0;
    for (int k0 = 0; k0 < I_DIM; k0 += 32, it ^= 1) {
        const int kn = k0 + 32;
        const bool more = (kn < I_DIM);
        float4 w0 = {}, w1 = {};
        if (more) {
            async_copy_b128(&as_[it ^ 1][m_ld * 32 + kk], acrow + kn + kk);
            const float4* s = reinterpret_cast<const float4*>(wrow + kn + kk);
            w0 = s[0]; w1 = s[1];
        }

        const unsigned short* ab = as_[it];
        const unsigned short* db = ds_[it];
        Frag16 fa, fb0, fb1;
        fa.u[0]  = *reinterpret_cast<const uint4*>(&ab[mrow * 32 + kbA]);
        fa.u[1]  = *reinterpret_cast<const uint4*>(&ab[mrow * 32 + kbA + 8]);
        fb0.u[0] = *reinterpret_cast<const uint4*>(&db[(nb0 + nl) * 32 + kbB]);
        fb0.u[1] = *reinterpret_cast<const uint4*>(&db[(nb0 + nl) * 32 + kbB + 8]);
        fb1.u[0] = *reinterpret_cast<const uint4*>(&db[(nb1 + nl) * 32 + kbB]);
        fb1.u[1] = *reinterpret_cast<const uint4*>(&db[(nb1 + nl) * 32 + kbB + 8]);

        a0 = __builtin_amdgcn_wmma_f32_16x16x32_bf16(false, fa.v, false, fb0.v, (short)0, a0, false, false);
        a1 = __builtin_amdgcn_wmma_f32_16x16x32_bf16(false, fa.v, false, fb1.v, (short)0, a1, false, false);

        if (more) {
            uint4 p;
            p.x = pack_bf16x2(w0.x, w0.y); p.y = pack_bf16x2(w0.z, w0.w);
            p.z = pack_bf16x2(w1.x, w1.y); p.w = pack_bf16x2(w1.z, w1.w);
            *reinterpret_cast<uint4*>(&ds_[it ^ 1][m_ld * 32 + kk]) = p;
            wait_async0();
        }
        __syncthreads();
    }

    const int mb = (lane < 16) ? 0 : 8;
#pragma unroll
    for (int half = 0; half < 2; ++half) {
        const v8f& a = half ? a1 : a0;
        int hg = hbase + nh * 32 + half * 16 + nl;
#pragma unroll
        for (int r = 0; r < 8; ++r) {
            int ml = mi * 16 + mb + r;
            if (tile * 64 + ml < cnt) {
                int tokn = toks[ml];
                float s = scores[tokn * E_NUM + e];
                atomicAdd(&out[(size_t)tokn * H_DIM + hg], s * a[r]);
            }
        }
    }
}

// ---------------------------------------------------------------------------
extern "C" void kernel_launch(void* const* d_in, const int* in_sizes, int n_in,
                              void* d_out, int out_size, void* d_ws, size_t ws_size,
                              hipStream_t stream)
{
    const float* x  = (const float*)d_in[0];
    const float* Wr = (const float*)d_in[1];
    const float* br = (const float*)d_in[2];
    const float* Wg = (const float*)d_in[3];
    const float* bg = (const float*)d_in[4];
    const float* Wu = (const float*)d_in[5];
    const float* bu = (const float*)d_in[6];
    const float* Wd = (const float*)d_in[7];
    const float* bd = (const float*)d_in[8];

    float* out    = (float*)d_out;
    float* scores = out + (size_t)N_TOK * H_DIM;   // tuple: [out | scores]

    char* ws = (char*)d_ws;
    int* counts  = (int*)(ws + 0);
    int* offsets = (int*)(ws + 32);
    int* tok_ids = (int*)(ws + 64);
    unsigned short* xb  = (unsigned short*)(ws + 33024);
    unsigned short* act = (unsigned short*)(ws + 33024 + (size_t)N_TOK * H_DIM * 2);

    router_kernel<<<(N_TOK * 32) / 256, 256, 0, stream>>>(x, Wr, br, scores, xb);
    compact_kernel<<<E_NUM, 32, 0, stream>>>(scores, counts, tok_ids);
    offsets_kernel<<<1, 32, 0, stream>>>(counts, offsets);
    bias_init_kernel<<<(N_TOK * H_DIM + 255) / 256, 256, 0, stream>>>(bd, scores, out);
    moe_gateup_kernel<<<dim3(I_DIM / 64, N_TOK / 64, E_NUM), 256, 0, stream>>>(
        xb, Wg, bg, Wu, bu, counts, offsets, tok_ids, act);
    moe_down_kernel<<<dim3(H_DIM / 64, N_TOK / 64, E_NUM), 256, 0, stream>>>(
        act, Wd, counts, offsets, tok_ids, scores, out);
}
